// GraphAttention_41369124995413
// MI455X (gfx1250) — matrix operand order repaired and checked by reading, not compiled, not run on hardware
//
#include <hip/hip_runtime.h>

#define B 4
#define N 4096
#define FIN 128
#define FOUT 64
#define ALPHA 0.2f

typedef __attribute__((ext_vector_type(16))) __bf16 v16bf;
typedef __attribute__((ext_vector_type(8)))  __bf16 bf16x8;
typedef __attribute__((ext_vector_type(8)))  float  v8f;

// ---------------------------------------------------------------------------
// K1: h = input @ weight ; f1 = h@a1 ; f2 = h@a2
// grid 64, block 256; one thread per (b,n) row.
// ---------------------------------------------------------------------------
__global__ void k1_linear(const float* __restrict__ in, const float* __restrict__ w,
                          const float* __restrict__ a1, const float* __restrict__ a2,
                          float* __restrict__ h, float* __restrict__ f1, float* __restrict__ f2) {
    __shared__ float wsh[FIN * FOUT];
    __shared__ float a1s[FOUT], a2s[FOUT];
    int t = threadIdx.x;
    for (int i = t; i < FIN * FOUT / 4; i += 256)
        ((float4*)wsh)[i] = ((const float4*)w)[i];
    if (t < FOUT) { a1s[t] = a1[t]; a2s[t] = a2[t]; }
    __syncthreads();

    int row = blockIdx.x * 256 + t;            // b*N + n
    const float* xr = in + (size_t)row * FIN;
    float acc[FOUT];
#pragma unroll
    for (int o = 0; o < FOUT; ++o) acc[o] = 0.0f;

    for (int k0 = 0; k0 < FIN; k0 += 4) {
        float4 x = *(const float4*)(xr + k0);
        float xs[4] = {x.x, x.y, x.z, x.w};
#pragma unroll
        for (int kk = 0; kk < 4; ++kk) {
            float xv = xs[kk];
            const float* wr = wsh + (k0 + kk) * FOUT;
#pragma unroll
            for (int o = 0; o < FOUT; o += 4) {
                float4 wv = *(const float4*)(wr + o);
                acc[o + 0] += xv * wv.x;
                acc[o + 1] += xv * wv.y;
                acc[o + 2] += xv * wv.z;
                acc[o + 3] += xv * wv.w;
            }
        }
    }
    float s1 = 0.0f, s2 = 0.0f;
#pragma unroll
    for (int o = 0; o < FOUT; ++o) { s1 += acc[o] * a1s[o]; s2 += acc[o] * a2s[o]; }
    float* hr = h + (size_t)row * FOUT;
#pragma unroll
    for (int o = 0; o < FOUT; o += 4) {
        float4 v = {acc[o], acc[o + 1], acc[o + 2], acc[o + 3]};
        *(float4*)(hr + o) = v;
    }
    f1[row] = s1;
    f2[row] = s2;
}

// ---------------------------------------------------------------------------
// K2a: column-sum partials  Sp[nc][b*N+m] = sum_{n in chunk} mask*exp(leaky(f1[n]+f2[m]))
// grid 1024 (= B * 16 mtiles * 16 nchunks), block 256.
// ---------------------------------------------------------------------------
__global__ void k2a_colsum(const int* __restrict__ adj, const float* __restrict__ f1,
                           const float* __restrict__ f2, float* __restrict__ Sp) {
    int bx = blockIdx.x;
    int b = bx >> 8;
    int rem = bx & 255;
    int mt = rem & 15;
    int nc = rem >> 4;
    int m = mt * 256 + threadIdx.x;

    float f2m = f2[b * N + m];
    const int* col = adj + (size_t)b * N * N + m;
    int nbase = nc * 256;
    float acc = 0.0f;
    for (int i = 0; i < 256; ++i) {
        float f1n = f1[b * N + nbase + i];           // uniform -> scalar load
        int a = col[(size_t)(nbase + i) * N];        // coalesced across threads
        float s = f1n + f2m;
        s = (s > 0.0f) ? s : ALPHA * s;
        float e = __expf(s);
        acc += (a > 0) ? e : 0.0f;
    }
    Sp[nc * (B * N) + b * N + m] = acc;
}

// ---------------------------------------------------------------------------
// K2b: reduce 16 partials -> rcpS = 1/S   (grid 64, block 256)
// ---------------------------------------------------------------------------
__global__ void k2b_rcp(const float* __restrict__ Sp, float* __restrict__ rcpS) {
    int i = blockIdx.x * 256 + threadIdx.x;          // < B*N
    float s = 0.0f;
#pragma unroll
    for (int k = 0; k < 16; ++k) s += Sp[k * (B * N) + i];
    rcpS[i] = 1.0f / s;
}

// ---------------------------------------------------------------------------
// K2c: hT[b][o][m] = bf16( h[b][m][o] * rcpS[b][m] )   (grid 256, block 256)
// 64x64 tile transpose via LDS (padded, conflict-free).
// ---------------------------------------------------------------------------
__global__ void k2c_transpose(const float* __restrict__ h, const float* __restrict__ rcpS,
                              __bf16* __restrict__ hT) {
    __shared__ float tile[64][65];
    int bx = blockIdx.x;
    int b = bx >> 6;
    int m0 = (bx & 63) * 64;
    int tx = threadIdx.x & 63;
    int ty = threadIdx.x >> 6;

    for (int r = ty; r < 64; r += 4) {
        int m = m0 + r;
        tile[r][tx] = h[(size_t)(b * N + m) * FOUT + tx] * rcpS[b * N + m];
    }
    __syncthreads();
    for (int r = ty; r < 64; r += 4) {
        int o = r;
        hT[(size_t)(b * FOUT + o) * N + m0 + tx] = (__bf16)tile[tx][o];
    }
}

// ---------------------------------------------------------------------------
// K3: h' partials via WMMA.  D'[o][n] = sum_m H~T[o][m] * P[m][n]
//   A (16o x 32m) = hT tiles, constant-offset b128 loads off ONE pointer,
//     shared across TWO n-tiles (halves A/L2 traffic per WMMA).
//   B (32m x 16n) = mask*exp(leaky(f1[n]+f2[m])), computed in registers.
// grid 256, block 256 (8 waves); wave -> (b, 32-row n-group, k-split of 1024).
// ---------------------------------------------------------------------------
__global__ void k3_wmma(const int* __restrict__ adj, const float* __restrict__ f1,
                        const float* __restrict__ f2, const __bf16* __restrict__ hT,
                        float* __restrict__ part) {
    int wid = (blockIdx.x * 256 + threadIdx.x) >> 5;   // 0..2047
    int lane = threadIdx.x & 31;
    int b = wid >> 9;           // 512 waves per batch
    int rem = wid & 511;
    int nt2 = rem >> 2;         // 0..127  (32-row group)
    int ks = rem & 3;           // k-split
    int n0 = nt2 * 32;
    int c = lane & 15;          // B/D column within tile, A row within tile
    int hh = lane >> 4;         // lane-half selector
    int mbeg = ks * 1024;

    int na = n0 + c;            // n-tile 0 row
    int nb = n0 + 16 + c;       // n-tile 1 row
    float f1a = f1[b * N + na];
    float f1b = f1[b * N + nb];

    // Incrementing pointers; everything else is constant instruction offsets.
    const __bf16* hp = hT + (size_t)(b * FOUT + c) * N + mbeg + 16 * hh; // +w*16*N per o-tile
    const int*    pA = adj + (size_t)(b * N + na) * N + mbeg + 16 * hh;
    const int*    pB = adj + (size_t)(b * N + nb) * N + mbeg + 16 * hh;
    const float*  pF = f2 + b * N + mbeg + 16 * hh;

    v8f acc[2][4];
#pragma unroll
    for (int t = 0; t < 2; ++t)
#pragma unroll
        for (int w = 0; w < 4; ++w) acc[t][w] = (v8f){0, 0, 0, 0, 0, 0, 0, 0};

#pragma unroll 1
    for (int it = 0; it < 32; ++it) {      // uniform trip count -> scalar loop
        // ---- issue all A-tile loads up front (latency covered by B build) ----
        bf16x8 a0l = *(const bf16x8*)(hp);
        bf16x8 a0h = *(const bf16x8*)(hp + 8);
        bf16x8 a1l = *(const bf16x8*)(hp + 16 * N);
        bf16x8 a1h = *(const bf16x8*)(hp + 16 * N + 8);
        bf16x8 a2l = *(const bf16x8*)(hp + 32 * N);
        bf16x8 a2h = *(const bf16x8*)(hp + 32 * N + 8);
        bf16x8 a3l = *(const bf16x8*)(hp + 48 * N);
        bf16x8 a3h = *(const bf16x8*)(hp + 48 * N + 8);
        // ---- shared f2 chunk + adj row A ----
        const float4* fp = (const float4*)pF;
        float4 q0 = fp[0], q1 = fp[1], q2 = fp[2], q3 = fp[3];
        const int4* apA = (const int4*)pA;
        int4 x0 = apA[0], x1 = apA[1], x2 = apA[2], x3 = apA[3];
        float fi[16] = {q0.x, q0.y, q0.z, q0.w, q1.x, q1.y, q1.z, q1.w,
                        q2.x, q2.y, q2.z, q2.w, q3.x, q3.y, q3.z, q3.w};
        int aiA[16] = {x0.x, x0.y, x0.z, x0.w, x1.x, x1.y, x1.z, x1.w,
                       x2.x, x2.y, x2.z, x2.w, x3.x, x3.y, x3.z, x3.w};
        v16bf BvA;
#pragma unroll
        for (int j = 0; j < 16; ++j) {
            float s = f1a + fi[j];
            s = (s > 0.0f) ? s : ALPHA * s;
            float e = __expf(s);
            e = (aiA[j] > 0) ? e : 0.0f;
            BvA[j] = (__bf16)e;
        }
        // ---- adj row B (after BvA build frees x regs) ----
        const int4* apB = (const int4*)pB;
        int4 y0 = apB[0], y1 = apB[1], y2 = apB[2], y3 = apB[3];
        int aiB[16] = {y0.x, y0.y, y0.z, y0.w, y1.x, y1.y, y1.z, y1.w,
                       y2.x, y2.y, y2.z, y2.w, y3.x, y3.y, y3.z, y3.w};
        v16bf BvB;
#pragma unroll
        for (int j = 0; j < 16; ++j) {
            float s = f1b + fi[j];
            s = (s > 0.0f) ? s : ALPHA * s;
            float e = __expf(s);
            e = (aiB[j] > 0) ? e : 0.0f;
            BvB[j] = (__bf16)e;
        }
        // ---- assemble A operands and run 8 WMMAs ----
        v16bf Av0, Av1, Av2, Av3;
#pragma unroll
        for (int e = 0; e < 8; ++e) {
            Av0[e] = a0l[e]; Av0[e + 8] = a0h[e];
            Av1[e] = a1l[e]; Av1[e + 8] = a1h[e];
            Av2[e] = a2l[e]; Av2[e + 8] = a2h[e];
            Av3[e] = a3l[e]; Av3[e + 8] = a3h[e];
        }
        acc[0][0] = __builtin_amdgcn_wmma_f32_16x16x32_bf16(false, Av0, false, BvA, (short)0, acc[0][0], false, false);
        acc[1][0] = __builtin_amdgcn_wmma_f32_16x16x32_bf16(false, Av0, false, BvB, (short)0, acc[1][0], false, false);
        acc[0][1] = __builtin_amdgcn_wmma_f32_16x16x32_bf16(false, Av1, false, BvA, (short)0, acc[0][1], false, false);
        acc[1][1] = __builtin_amdgcn_wmma_f32_16x16x32_bf16(false, Av1, false, BvB, (short)0, acc[1][1], false, false);
        acc[0][2] = __builtin_amdgcn_wmma_f32_16x16x32_bf16(false, Av2, false, BvA, (short)0, acc[0][2], false, false);
        acc[1][2] = __builtin_amdgcn_wmma_f32_16x16x32_bf16(false, Av2, false, BvB, (short)0, acc[1][2], false, false);
        acc[0][3] = __builtin_amdgcn_wmma_f32_16x16x32_bf16(false, Av3, false, BvA, (short)0, acc[0][3], false, false);
        acc[1][3] = __builtin_amdgcn_wmma_f32_16x16x32_bf16(false, Av3, false, BvB, (short)0, acc[1][3], false, false);

        hp += 32; pA += 32; pB += 32; pF += 32;
    }

    // D'[o][n]: VGPR v -> o = 16w + v + 8*hh ; col -> n
#pragma unroll
    for (int t = 0; t < 2; ++t) {
        int n = (t == 0) ? na : nb;
        float* pb = part + ((size_t)((ks * B + b) * N + n)) * FOUT + 8 * hh;
#pragma unroll
        for (int w = 0; w < 4; ++w)
#pragma unroll
            for (int v = 0; v < 8; ++v) pb[16 * w + v] = acc[t][w][v];
    }
}

// ---------------------------------------------------------------------------
// K4: sum 4 K-split partials + ELU -> out   (grid 4096, block 256)
// ---------------------------------------------------------------------------
__global__ void k4_reduce_elu(const float* __restrict__ part, float* __restrict__ out) {
    int idx = blockIdx.x * 256 + threadIdx.x;        // < B*N*FOUT
    const int St = B * N * FOUT;
    float x = part[idx] + part[idx + St] + part[idx + 2 * St] + part[idx + 3 * St];
    out[idx] = (x > 0.0f) ? x : (__expf(x) - 1.0f);
}

// ---------------------------------------------------------------------------
extern "C" void kernel_launch(void* const* d_in, const int* in_sizes, int n_in,
                              void* d_out, int out_size, void* d_ws, size_t ws_size,
                              hipStream_t stream) {
    const float* input  = (const float*)d_in[0];
    const float* weight = (const float*)d_in[1];
    const float* a1     = (const float*)d_in[2];
    const float* a2     = (const float*)d_in[3];
    const int*   adj    = (const int*)d_in[4];
    float* out = (float*)d_out;

    // workspace layout (floats)
    float* wsf  = (float*)d_ws;
    float* Hf   = wsf;                                   // B*N*64      = 1,048,576
    float* F1   = Hf   + (size_t)B * N * FOUT;           // B*N         = 16,384
    float* F2   = F1   + (size_t)B * N;
    float* RcpS = F2   + (size_t)B * N;
    float* Sp   = RcpS + (size_t)B * N;                  // 16*B*N      = 262,144
    float* Part = Sp   + (size_t)16 * B * N;             // 4*B*N*64    = 4,194,304
    __bf16* HT  = (__bf16*)(Part + (size_t)4 * B * N * FOUT);   // B*64*N bf16 (16B aligned)

    k1_linear<<<(B * N) / 256, 256, 0, stream>>>(input, weight, a1, a2, Hf, F1, F2);
    k2a_colsum<<<B * 16 * 16, 256, 0, stream>>>(adj, F1, F2, Sp);
    k2b_rcp<<<(B * N) / 256, 256, 0, stream>>>(Sp, RcpS);
    k2c_transpose<<<B * (N / 64), 256, 0, stream>>>(Hf, RcpS, HT);
    k3_wmma<<<256, 256, 0, stream>>>(adj, F1, F2, HT, Part);
    k4_reduce_elu<<<(B * N * FOUT) / 256, 256, 0, stream>>>(Part, out);
}
